// TripletLossSscl_17686675325025
// MI455X (gfx1250) — matrix operand order repaired and checked by reading, not compiled, not run on hardware
//
#include <hip/hip_runtime.h>
#include <hip/hip_bf16.h>

typedef __bf16 v8bf  __attribute__((ext_vector_type(8)));
typedef __bf16 v16bf __attribute__((ext_vector_type(16)));
typedef float  v8f   __attribute__((ext_vector_type(8)));

#define NROWS    8192
#define KDIM     2048
#define MT       32            // rows per block in sim_mine
#define NCHUNK   2048          // columns per block (grid.y = NROWS/NCHUNK)
#define NSWEEP   512           // columns per sweep iter (8 waves * 64)
#define LSTR     (KDIM + 8)    // LDS row stride in bf16 elems
#define MARGIN_F 0.3f
#define EPS_F    1e-12f
#define INF_BITS 0x7f800000u

// ---------------------------------------------------------------------------
// Kernel 1: L2-normalize each row to bf16; init per-row min/max bit arrays.
// ---------------------------------------------------------------------------
__global__ __launch_bounds__(256) void normalize_rows(
    const float* __restrict__ in, unsigned short* __restrict__ xn_u16,
    unsigned* __restrict__ ap, unsigned* __restrict__ an) {
  const int row = blockIdx.x;
  const int tid = threadIdx.x;
  __bf16* xn = (__bf16*)xn_u16;

  const float4* ip = (const float4*)(in + (size_t)row * KDIM);
  float4 v0 = ip[tid * 2 + 0];
  float4 v1 = ip[tid * 2 + 1];
  float ss = v0.x * v0.x + v0.y * v0.y + v0.z * v0.z + v0.w * v0.w +
             v1.x * v1.x + v1.y * v1.y + v1.z * v1.z + v1.w * v1.w;
#pragma unroll
  for (int off = 16; off >= 1; off >>= 1) ss += __shfl_xor(ss, off, 32);

  __shared__ float wred[8];
  __shared__ float stot;
  if ((tid & 31) == 0) wred[tid >> 5] = ss;
  __syncthreads();
  if (tid == 0) {
    float t = 0.f;
#pragma unroll
    for (int i = 0; i < 8; ++i) t += wred[i];
    stot = t;
    ap[row] = INF_BITS;  // running min over positives (+inf)
    an[row] = 0u;        // running max over negatives (sims >= EPS > 0)
  }
  __syncthreads();

  const float inv = 1.0f / fmaxf(sqrtf(stot), EPS_F);
  const float f[8] = {v0.x, v0.y, v0.z, v0.w, v1.x, v1.y, v1.z, v1.w};
  v8bf o;
#pragma unroll
  for (int i = 0; i < 8; ++i) o[i] = (__bf16)(f[i] * inv);
  *(v8bf*)(xn + (size_t)row * KDIM + tid * 8) = o;
}

// A-tile fetch from LDS, 16x32 bf16, in contiguous k order (same order as load_b):
// lane lo = row M; VGPR0-3 hold K = hi*16+0..7, VGPR4-7 hold K = hi*16+8..15.
__device__ __forceinline__ v16bf load_a(const __bf16* row, int k0, int hi) {
  v8bf a_lo = *(const v8bf*)(row + k0 + hi * 16);
  v8bf a_hi = *(const v8bf*)(row + k0 + hi * 16 + 8);
  return __builtin_shufflevector(a_lo, a_hi, 0, 1, 2, 3, 4, 5, 6, 7, 8, 9, 10,
                                 11, 12, 13, 14, 15);
}

// B-tile fetch: SGPR base (xn) + 32-bit per-lane element offset -> GVS saddr
// addressing with the k-offset folded into the 24-bit instruction immediate.
__device__ __forceinline__ v16bf load_b(const __bf16* __restrict__ xn,
                                        unsigned boff, int koff) {
  return *(const v16bf*)(xn + (boff + (unsigned)koff));
}

__device__ __forceinline__ v8f wmma_bf16(v16bf a, v16bf b, v8f c) {
  return __builtin_amdgcn_wmma_f32_16x16x32_bf16(false, a, false, b, (short)0,
                                                 c, false, false);
}

// ---------------------------------------------------------------------------
// Kernel 2: fused Gram-matrix (X * X^T) + hard mining.
// 8 waves/block; each wave owns a 32x64 C macro-tile (2M x 4N WMMA tiles).
// K loop is ping-pong pipelined (two named tile register sets, no rotation
// copies): while the WMMAs consume one set, loads fill the other.
// ---------------------------------------------------------------------------
__global__ __launch_bounds__(256) void sim_mine(
    const unsigned short* __restrict__ xn_u16, const int* __restrict__ lab32,
    unsigned* __restrict__ ap, unsigned* __restrict__ an) {
  extern __shared__ char smem_raw[];
  __bf16* As = (__bf16*)smem_raw;
  const __bf16* xn = (const __bf16*)xn_u16;

  const int tid   = threadIdx.x;
  const int lane  = tid & 31;
  const int wave  = tid >> 5;    // 0..7 : N sub-block within sweep
  const int hi    = lane >> 4;   // 0/1  : K half (A/B), M+8 (C)
  const int lo    = lane & 15;   // M row (A), N col (B, C)
  const int m0    = blockIdx.x * MT;
  const int nbase = blockIdx.y * NCHUNK;

  // Stage the 32 x 2048 A strip into LDS (16B units, coalesced).
  for (int u = tid; u < MT * (KDIM / 8); u += 256) {
    const int r  = u >> 8;   // KDIM/8 == 256 units per row
    const int c8 = u & 255;
    *(v8bf*)(As + r * LSTR + c8 * 8) =
        *(const v8bf*)(xn + (size_t)(m0 + r) * KDIM + c8 * 8);
  }
  __syncthreads();

  // Row labels (int64 values < 128 -> low dword suffices) for both M tiles.
  int rowlab[2][8];
#pragma unroll
  for (int mi = 0; mi < 2; ++mi)
#pragma unroll
    for (int r = 0; r < 8; ++r)
      rowlab[mi][r] = lab32[2 * (m0 + mi * 16 + r + 8 * hi)];

  float minp[2][8], maxn[2][8];
#pragma unroll
  for (int mi = 0; mi < 2; ++mi)
#pragma unroll
    for (int r = 0; r < 8; ++r) {
      minp[mi][r] = __builtin_inff();
      maxn[mi][r] = 0.0f;
    }

  const __bf16* arow0 = As + lo * LSTR;         // M tile 0: rows m0+0..15
  const __bf16* arow1 = As + (16 + lo) * LSTR;  // M tile 1: rows m0+16..31

  for (int n0 = nbase + wave * 64; n0 < nbase + NCHUNK; n0 += NSWEEP) {
    // B 32x16 layout: lane lo = col N; lanes 0-15 K=0..15, 16-31 K=16..31
    // -> 32B contiguous reads of row (n0+16*ni+lo) of X (L2-resident).
    int collab[4];
    unsigned boff[4];  // element offsets from xn (fits 32 bits: < 16M elems)
#pragma unroll
    for (int ni = 0; ni < 4; ++ni) {
      collab[ni] = lab32[2 * (n0 + 16 * ni + lo)];
      boff[ni] = (unsigned)(n0 + 16 * ni + lo) * KDIM + hi * 16;
    }
    // Prefetch next sweep's B rows into near caches (high locality).
    if (n0 + NSWEEP < nbase + NCHUNK) {
      __builtin_prefetch((const void*)(xn + boff[0] + (size_t)NSWEEP * KDIM),
                         0, 3);
      __builtin_prefetch((const void*)(xn + boff[2] + (size_t)NSWEEP * KDIM),
                         0, 3);
    }

    v8f c[2][4];
#pragma unroll
    for (int mi = 0; mi < 2; ++mi)
#pragma unroll
      for (int ni = 0; ni < 4; ++ni) c[mi][ni] = (v8f){};

    // Ping-pong pipeline: set P holds tile k0, set Q holds tile k0+32.
    v16bf aP0 = load_a(arow0, 0, hi);
    v16bf aP1 = load_a(arow1, 0, hi);
    v16bf bP[4];
#pragma unroll
    for (int ni = 0; ni < 4; ++ni) bP[ni] = load_b(xn, boff[ni], 0);

    v16bf aQ0, aQ1, bQ[4];

#pragma unroll 1
    for (int k0 = 0; k0 < KDIM - 64; k0 += 64) {
      // Load tile k0+32 into Q while computing tile k0 from P.
      aQ0 = load_a(arow0, k0 + 32, hi);
      aQ1 = load_a(arow1, k0 + 32, hi);
#pragma unroll
      for (int ni = 0; ni < 4; ++ni) bQ[ni] = load_b(xn, boff[ni], k0 + 32);
#pragma unroll
      for (int ni = 0; ni < 4; ++ni) {
        c[0][ni] = wmma_bf16(aP0, bP[ni], c[0][ni]);
        c[1][ni] = wmma_bf16(aP1, bP[ni], c[1][ni]);
      }
      // Load tile k0+64 into P while computing tile k0+32 from Q.
      aP0 = load_a(arow0, k0 + 64, hi);
      aP1 = load_a(arow1, k0 + 64, hi);
#pragma unroll
      for (int ni = 0; ni < 4; ++ni) bP[ni] = load_b(xn, boff[ni], k0 + 64);
#pragma unroll
      for (int ni = 0; ni < 4; ++ni) {
        c[0][ni] = wmma_bf16(aQ0, bQ[ni], c[0][ni]);
        c[1][ni] = wmma_bf16(aQ1, bQ[ni], c[1][ni]);
      }
    }
    // Tail: P holds tile KDIM-64; load KDIM-32 into Q, compute both.
    aQ0 = load_a(arow0, KDIM - 32, hi);
    aQ1 = load_a(arow1, KDIM - 32, hi);
#pragma unroll
    for (int ni = 0; ni < 4; ++ni) bQ[ni] = load_b(xn, boff[ni], KDIM - 32);
#pragma unroll
    for (int ni = 0; ni < 4; ++ni) {
      c[0][ni] = wmma_bf16(aP0, bP[ni], c[0][ni]);
      c[1][ni] = wmma_bf16(aP1, bP[ni], c[1][ni]);
    }
#pragma unroll
    for (int ni = 0; ni < 4; ++ni) {
      c[0][ni] = wmma_bf16(aQ0, bQ[ni], c[0][ni]);
      c[1][ni] = wmma_bf16(aQ1, bQ[ni], c[1][ni]);
    }

    // Mining epilogue: C VGPR r / lane -> (M = r + 8*hi, N = lo).
#pragma unroll
    for (int mi = 0; mi < 2; ++mi)
#pragma unroll
      for (int ni = 0; ni < 4; ++ni)
#pragma unroll
        for (int r = 0; r < 8; ++r) {
          const float s = fmaxf(c[mi][ni][r], EPS_F);
          const bool same = (rowlab[mi][r] == collab[ni]);
          minp[mi][r] = same ? fminf(minp[mi][r], s) : minp[mi][r];
          maxn[mi][r] = same ? maxn[mi][r] : fmaxf(maxn[mi][r], s);
        }
  }

  // Cross-lane / cross-wave reduction: positive floats order as uints.
  __syncthreads();  // A strip dead; reuse LDS for 2*MT uint slots
  unsigned* redmin = (unsigned*)smem_raw;
  unsigned* redmax = redmin + MT;
  if (tid < MT) { redmin[tid] = INF_BITS; redmax[tid] = 0u; }
  __syncthreads();

#pragma unroll
  for (int mi = 0; mi < 2; ++mi)
#pragma unroll
    for (int r = 0; r < 8; ++r) {
      float mn = minp[mi][r], mx = maxn[mi][r];
#pragma unroll
      for (int off = 8; off >= 1; off >>= 1) {
        mn = fminf(mn, __shfl_xor(mn, off, 16));
        mx = fmaxf(mx, __shfl_xor(mx, off, 16));
      }
      if (lo == 0) {
        const int rr = mi * 16 + r + 8 * hi;
        atomicMin(&redmin[rr], __float_as_uint(mn));
        atomicMax(&redmax[rr], __float_as_uint(mx));
      }
    }
  __syncthreads();
  if (tid < MT) {  // merge across the grid.y N-chunks
    atomicMin(&ap[m0 + tid], redmin[tid]);
    atomicMax(&an[m0 + tid], redmax[tid]);
  }
}

// ---------------------------------------------------------------------------
// Kernel 3: loss = mean(max(0, sim_an - sim_ap + margin)); empty pos/neg sets
// behave like the reference's +/-inf fills (term -> 0).
// ---------------------------------------------------------------------------
__global__ __launch_bounds__(256) void finalize_loss(
    const unsigned* __restrict__ ap, const unsigned* __restrict__ an,
    float* __restrict__ out) {
  __shared__ float red[256];
  float s = 0.f;
  for (int i = threadIdx.x; i < NROWS; i += 256) {
    const unsigned a = ap[i], b = an[i];
    if (a != INF_BITS && b != 0u)
      s += fmaxf(__uint_as_float(b) - __uint_as_float(a) + MARGIN_F, 0.0f);
  }
  red[threadIdx.x] = s;
  __syncthreads();
  for (int off = 128; off > 0; off >>= 1) {
    if (threadIdx.x < off) red[threadIdx.x] += red[threadIdx.x + off];
    __syncthreads();
  }
  if (threadIdx.x == 0) out[0] = red[0] / (float)NROWS;
}

// ---------------------------------------------------------------------------
extern "C" void kernel_launch(void* const* d_in, const int* in_sizes, int n_in,
                              void* d_out, int out_size, void* d_ws,
                              size_t ws_size, hipStream_t stream) {
  (void)in_sizes; (void)n_in; (void)out_size; (void)ws_size;
  const float* inputs = (const float*)d_in[0];
  const int* targets32 = (const int*)d_in[1];  // int64 labels, values < 128
  float* out = (float*)d_out;

  // Workspace layout: [ bf16 Xn : 32 MB ][ ap bits : 32 KB ][ an bits : 32 KB ]
  unsigned short* xn = (unsigned short*)d_ws;
  unsigned* ap =
      (unsigned*)((char*)d_ws + (size_t)NROWS * KDIM * sizeof(unsigned short));
  unsigned* an = ap + NROWS;

  normalize_rows<<<NROWS, 256, 0, stream>>>(inputs, xn, ap, an);

  const size_t lds_bytes = (size_t)MT * LSTR * sizeof(unsigned short);
  dim3 grid(NROWS / MT, NROWS / NCHUNK);
  sim_mine<<<grid, 256, lds_bytes, stream>>>(xn, targets32, ap, an);

  finalize_loss<<<1, 256, 0, stream>>>(ap, an, out);
}
